// BahdanauAttention_8169027797047
// MI455X (gfx1250) — compile-verified, matrix operand order checked by reading, not run on hardware
//
#include <hip/hip_runtime.h>
#include <math.h>

// Bahdanau attention fused for MI455X (gfx1250, wave32, WMMA bf16 + async-LDS).
// B=64, L=1024, D=H=U=1024.

#define B_ 64
#define L_ 1024
#define D_ 1024
#define U_ 1024

#define SA_STRIDE 1032          // 1024 + 8 bf16 pad: lane stride = 516 banks = 4 mod 64
#define SB_STRIDE 40            // 32 + 8 bf16 pad (80 B rows): 20 banks/lane, 16B aligned

typedef __attribute__((ext_vector_type(16))) __bf16 v16bf;
typedef __attribute__((ext_vector_type(8)))  float  v8f;
typedef __attribute__((ext_vector_type(4)))  int    v4i;

typedef __attribute__((address_space(1))) v4i* g128p;   // global 16B chunk
typedef __attribute__((address_space(3))) v4i* l128p;   // LDS    16B chunk

struct Frag32 { uint4 lo, hi; };   // 32 bytes == one v16bf fragment

static __device__ __forceinline__ unsigned short f2bf(float f) {
    unsigned int x = __float_as_uint(f);
    x += 0x7fffu + ((x >> 16) & 1u);          // round-to-nearest-even
    return (unsigned short)(x >> 16);
}

// ---- CDNA5 async global->LDS copy (ASYNCcnt-tracked DMA, no VGPR staging) ----
static __device__ __forceinline__ void async_g2l_b128(const void* g, void* l) {
#if __has_builtin(__builtin_amdgcn_global_load_async_to_lds_b128)
    __builtin_amdgcn_global_load_async_to_lds_b128((g128p)g, (l128p)l, 0, 0);
#else
    unsigned lds = (unsigned)(unsigned long long)(size_t)l;
    unsigned long long ga = (unsigned long long)(size_t)g;
    asm volatile("global_load_async_to_lds_b128 %0, %1, off" :: "v"(lds), "v"(ga) : "memory");
#endif
}

#if __has_builtin(__builtin_amdgcn_s_wait_asynccnt)
#define WAITA(n) __builtin_amdgcn_s_wait_asynccnt(n)
#else
#define WAITA(n) asm volatile("s_wait_asynccnt " #n ::: "memory")
#endif

// ---------------------------------------------------------------------------
// 0) init: zero context output, seed score buffer with bV (folds final bias)
// ---------------------------------------------------------------------------
__global__ __launch_bounds__(256) void k_init(float* ctx, float* score, const float* bV) {
    int i = blockIdx.x * 256 + threadIdx.x;   // 65536 = B*D = B*L
    ctx[i]   = 0.0f;
    score[i] = bV[0];
}

// ---------------------------------------------------------------------------
// 1) W1 [D,U] fp32 -> W1T [U,D] bf16 (B-fragments become contiguous in K)
// ---------------------------------------------------------------------------
__global__ __launch_bounds__(256) void k_w1t(const float* __restrict__ W1,
                                             unsigned short* __restrict__ w1t) {
    int idx = blockIdx.x * 256 + threadIdx.x;     // 1M elements, coalesced read
    int k = idx >> 10;
    int n = idx & 1023;
    w1t[(size_t)n * D_ + k] = f2bf(W1[idx]);
}

// ---------------------------------------------------------------------------
// 2) proj_h[b][u] = hidden[b,:] @ W2[:,u] + b1[u] + b2[u]   (b1 folded here)
// ---------------------------------------------------------------------------
__global__ __launch_bounds__(1024) void k_projh(const float* __restrict__ hidden,
                                                const float* __restrict__ W2,
                                                const float* __restrict__ b1,
                                                const float* __restrict__ b2,
                                                float* __restrict__ projh) {
    int b = blockIdx.x, u = threadIdx.x;
    __shared__ float h[1024];
    h[u] = hidden[b * 1024 + u];
    __syncthreads();
    float sum = b1[u] + b2[u];
    #pragma unroll 4
    for (int k = 0; k < 1024; ++k)
        sum = fmaf(h[k], W2[(size_t)k * U_ + u], sum);
    projh[b * U_ + u] = sum;
}

// ---------------------------------------------------------------------------
// 3) Fused GEMM + tanh + dot(V).
//    M-tile = 32 rows / workgroup (2048 WGs, 8 waves). A staged bf16 in LDS.
//    B (bf16 W1T) triple-buffered in LDS via async DMA (K=32 slabs, depth-2
//    prefetch). Each wave: 8 N-tiles x 2 M-subtiles -> 16 f32 accumulators,
//    B fragment reused across both M-subtiles. Bank-conflict-free padding.
// ---------------------------------------------------------------------------
__device__ __forceinline__ void issue_slab(const unsigned short* __restrict__ w1t,
                                           unsigned short* sbuf, int kb, int tid) {
    // slab: 1024 rows x 64 B data (K=32 bf16) in 80 B padded rows
    #pragma unroll
    for (int i = 0; i < 16; ++i) {
        int c = i * 256 + tid;            // 0..4095 16B-chunks
        int n = c >> 2;
        int p = c & 3;
        const void* g = (const char*)w1t + (size_t)n * 2048 + (size_t)kb * 64 + p * 16;
        void* l = (char*)sbuf + n * (SB_STRIDE * 2) + p * 16;
        async_g2l_b128(g, l);
    }
}

__device__ __forceinline__ void compute_slab(const unsigned short* sA,
                                             const unsigned short* sb,
                                             int kb, int wave, int m0, int koff,
                                             v8f acc[2][8]) {
    // Two A fragments (M-subtiles 0/1), each reused across 8 WMMAs
    Frag32 a0f, a1f;
    const uint4* pa0 = (const uint4*)&sA[m0 * SA_STRIDE + kb * 32 + koff];
    a0f.lo = pa0[0]; a0f.hi = pa0[2];
    const uint4* pa1 = (const uint4*)&sA[(16 + m0) * SA_STRIDE + kb * 32 + koff];
    a1f.lo = pa1[0]; a1f.hi = pa1[2];
    v16bf a0 = __builtin_bit_cast(v16bf, a0f);
    v16bf a1 = __builtin_bit_cast(v16bf, a1f);

    #pragma unroll
    for (int j = 0; j < 8; ++j) {
        int n0 = (wave * 8 + j) * 16 + m0;
        Frag32 bfr;                            // B frag reused for both subtiles
        const uint4* pb = (const uint4*)&sb[n0 * SB_STRIDE + koff];
        bfr.lo = pb[0]; bfr.hi = pb[2];
        v16bf bv = __builtin_bit_cast(v16bf, bfr);
        acc[0][j] = __builtin_amdgcn_wmma_f32_16x16x32_bf16(
            false, a0, false, bv, (short)0, acc[0][j], false, false);
        acc[1][j] = __builtin_amdgcn_wmma_f32_16x16x32_bf16(
            false, a1, false, bv, (short)0, acc[1][j], false, false);
    }
}

__global__ __launch_bounds__(256) void k_score(const float* __restrict__ feat,
                                               const unsigned short* __restrict__ w1t,
                                               const float* __restrict__ projh,
                                               const float* __restrict__ V,
                                               float* __restrict__ score) {
    __shared__ unsigned short sA[32 * SA_STRIDE];      // ~64.5 KB bf16 A tile
    __shared__ unsigned short sB[3][1024 * SB_STRIDE]; // 3 x 80 KB B slabs

    const int row0 = blockIdx.x * 32;             // global row in [0, B*L)
    const int bidx = row0 >> 10;                  // batch (tiles never straddle)
    const int tid  = threadIdx.x;
    const int lane = tid & 31;
    const int wave = tid >> 5;
    const int m0   = lane & 15;
    const int koff = (lane >> 4) * 8;             // A/B layout: hi lanes +8 in K

    // Prefetch first two B slabs via async DMA, overlapping with A conversion.
    issue_slab(w1t, sB[0], 0, tid);
    issue_slab(w1t, sB[1], 1, tid);

    // Stage 32 feature rows fp32 -> bf16 (float4 loads, uint2 LDS stores)
    const float4* fsrc = (const float4*)(feat + (size_t)row0 * D_);
    for (int i = tid; i < 32 * D_ / 4; i += 256) {
        int r = i >> 8;                            // 256 float4 per row
        int c = (i & 255) << 2;
        float4 f = fsrc[i];
        uint2 u;
        u.x = (unsigned)f2bf(f.x) | ((unsigned)f2bf(f.y) << 16);
        u.y = (unsigned)f2bf(f.z) | ((unsigned)f2bf(f.w) << 16);
        *(uint2*)&sA[r * SA_STRIDE + c] = u;
    }

    v8f acc[2][8];
    #pragma unroll
    for (int s = 0; s < 2; ++s)
        #pragma unroll
        for (int j = 0; j < 8; ++j) acc[s][j] = 0.0f;

    // Steady state: branch-free issue + wait + compute
    for (int kb = 0; kb < 30; ++kb) {
        __syncthreads();                           // buf[(kb+2)%3] free again
        issue_slab(w1t, sB[(kb + 2) % 3], kb + 2, tid);
        WAITA(32);                                 // slab kb complete (in-order)
        __syncthreads();                           // visible to all waves
        compute_slab(sA, sB[kb % 3], kb, wave, m0, koff, acc);
    }
    // Tail: kb = 30, 31 (nothing left to issue)
    __syncthreads();
    WAITA(16);
    __syncthreads();
    compute_slab(sA, sB[30 % 3], 30, wave, m0, koff, acc);
    __syncthreads();
    WAITA(0);
    __syncthreads();
    compute_slab(sA, sB[31 % 3], 31, wave, m0, koff, acc);

    // Epilogue: C layout -> lane = N (mod 16), VGPR i -> M = i + 8*(lane>=16)
    #pragma unroll
    for (int st = 0; st < 2; ++st) {
        float rowacc[8];
        #pragma unroll
        for (int i = 0; i < 8; ++i) rowacc[i] = 0.0f;

        #pragma unroll
        for (int j = 0; j < 8; ++j) {
            int n = (wave * 8 + j) * 16 + m0;
            float ph = projh[bidx * U_ + n];
            float vv = V[n];
            #pragma unroll
            for (int i = 0; i < 8; ++i)
                rowacc[i] += tanhf(acc[st][j][i] + ph) * vv;
        }

        #pragma unroll
        for (int i = 0; i < 8; ++i) {
            float s = rowacc[i];
            s += __shfl_xor(s, 1);                 // reduce within 16-lane halves
            s += __shfl_xor(s, 2);
            s += __shfl_xor(s, 4);
            s += __shfl_xor(s, 8);
            if (m0 == 0) {
                int m = (lane >> 4) * 8 + i;
                atomicAdd(&score[row0 + st * 16 + m], s);
            }
        }
    }
}

// ---------------------------------------------------------------------------
// 4) softmax over L per batch
// ---------------------------------------------------------------------------
__global__ __launch_bounds__(256) void k_softmax(const float* __restrict__ score,
                                                 float* __restrict__ attn) {
    int b = blockIdx.x, tid = threadIdx.x;
    __shared__ float red[256];
    const float* s = score + (size_t)b * L_;
    float v0 = s[tid], v1 = s[tid + 256], v2 = s[tid + 512], v3 = s[tid + 768];

    float m = fmaxf(fmaxf(v0, v1), fmaxf(v2, v3));
    red[tid] = m;
    __syncthreads();
    for (int o = 128; o > 0; o >>= 1) {
        if (tid < o) red[tid] = fmaxf(red[tid], red[tid + o]);
        __syncthreads();
    }
    m = red[0];
    __syncthreads();

    float e0 = expf(v0 - m), e1 = expf(v1 - m), e2 = expf(v2 - m), e3 = expf(v3 - m);
    red[tid] = e0 + e1 + e2 + e3;
    __syncthreads();
    for (int o = 128; o > 0; o >>= 1) {
        if (tid < o) red[tid] += red[tid + o];
        __syncthreads();
    }
    float inv = 1.0f / red[0];

    float* a = attn + (size_t)b * L_;
    a[tid]       = e0 * inv;
    a[tid + 256] = e1 * inv;
    a[tid + 512] = e2 * inv;
    a[tid + 768] = e3 * inv;
}

// ---------------------------------------------------------------------------
// 5) context[b][d] = sum_l attn[b][l] * features[b][l][d]  (bandwidth-bound)
//    float4 loads (global_load_b128), 512 blocks, L split 8-way + atomics.
// ---------------------------------------------------------------------------
__global__ __launch_bounds__(256) void k_context(const float* __restrict__ feat,
                                                 const float* __restrict__ attn,
                                                 float* __restrict__ ctx) {
    int b  = blockIdx.x;
    int l0 = blockIdx.y * 128;
    int tid = threadIdx.x;

    __shared__ float w[128];
    if (tid < 128) w[tid] = attn[(size_t)b * L_ + l0 + tid];
    __syncthreads();

    const float4* fp = (const float4*)(feat + ((size_t)b * L_ + l0) * D_) + tid;
    float4 s = make_float4(0.f, 0.f, 0.f, 0.f);
    #pragma unroll 2
    for (int j = 0; j < 128; ++j) {
        float4 f = fp[(size_t)j * (D_ / 4)];
        float wj = w[j];
        s.x = fmaf(wj, f.x, s.x);
        s.y = fmaf(wj, f.y, s.y);
        s.z = fmaf(wj, f.z, s.z);
        s.w = fmaf(wj, f.w, s.w);
    }
    float* c = ctx + b * D_ + tid * 4;
    atomicAdd(c + 0, s.x);
    atomicAdd(c + 1, s.y);
    atomicAdd(c + 2, s.z);
    atomicAdd(c + 3, s.w);
}

// ---------------------------------------------------------------------------
extern "C" void kernel_launch(void* const* d_in, const int* in_sizes, int n_in,
                              void* d_out, int out_size, void* d_ws, size_t ws_size,
                              hipStream_t stream) {
    const float* feat   = (const float*)d_in[0];   // [B, L, D]
    const float* hidden = (const float*)d_in[1];   // [B, H]
    const float* W1     = (const float*)d_in[2];   // [D, U]
    const float* b1     = (const float*)d_in[3];   // [U]
    const float* W2     = (const float*)d_in[4];   // [H, U]
    const float* b2     = (const float*)d_in[5];   // [U]
    const float* V      = (const float*)d_in[6];   // [U, 1]
    const float* bV     = (const float*)d_in[7];   // [1]

    float* out  = (float*)d_out;
    float* ctx  = out;                 // [B, D]   first 65536
    float* attn = out + B_ * D_;       // [B, L,1] next 65536

    // workspace: [0,2MB) W1T bf16 | [2MB,+256KB) proj_h | next 256KB score
    unsigned short* w1t = (unsigned short*)d_ws;
    float* projh = (float*)((char*)d_ws + (size_t)(2 << 20));
    float* score = (float*)((char*)d_ws + (size_t)(2 << 20) + (256 << 10));

    k_init   <<<dim3(256),      dim3(256),  0, stream>>>(ctx, score, bV);
    k_w1t    <<<dim3(4096),     dim3(256),  0, stream>>>(W1, w1t);
    k_projh  <<<dim3(64),       dim3(1024), 0, stream>>>(hidden, W2, b1, b2, projh);
    k_score  <<<dim3(2048),     dim3(256),  0, stream>>>(feat, w1t, projh, V, score);
    k_softmax<<<dim3(64),       dim3(256),  0, stream>>>(score, attn);
    k_context<<<dim3(64, 8),    dim3(256),  0, stream>>>(feat, attn, ctx);
}
